// BiLSTM_CRF_MODIFY_PARALLEL_5617817223776
// MI455X (gfx1250) — compile-verified, matrix operand order checked by reading, not run on hardware
//
#include <hip/hip_runtime.h>
#include <math.h>

// ---------------- problem constants (match reference) ----------------
#define S_   2048
#define E_   256
#define Hh_  256
#define G_   1024          // 4*Hh
#define T_   7
#define START_ 4
#define STOP_  5
#define NEG_ (-10000.0f)

// ---------------- vector types for WMMA ----------------
typedef __attribute__((ext_vector_type(8)))  __bf16 v8bf;
typedef __attribute__((ext_vector_type(16))) __bf16 v16bf;
typedef __attribute__((ext_vector_type(8)))  float  v8f;

// float -> bf16 (round-to-nearest-even) without relying on __bf16 scalar conversions
static __device__ __forceinline__ unsigned short f2bf(float f) {
  unsigned u = __float_as_uint(f);
  u += 0x7FFFu + ((u >> 16) & 1u);
  return (unsigned short)(u >> 16);
}
static __device__ __forceinline__ float sigmf(float x) {
  return 1.0f / (1.0f + __expf(-x));
}

// ---------------- workspace layout (bytes) ----------------
// total ~40 MB; entire working set is L2-resident on MI455X (192 MB L2)
static constexpr size_t OFF_XS   = 0;                                   // S*E bf16      (1 MB)
static constexpr size_t OFF_W    = OFF_XS  + (size_t)S_*E_*2;           // 2*G*E bf16    (1 MB)
static constexpr size_t OFF_GX   = OFF_W   + (size_t)2*G_*E_*2;         // 2*S*G f32     (16 MB) ; reused as path[t][s]
static constexpr size_t OFF_H    = OFF_GX  + (size_t)2*S_*G_*4;         // 2*S*Hh f32    (4 MB)
static constexpr size_t OFF_FA   = OFF_H   + (size_t)2*S_*Hh_*4;        // S*T f32
static constexpr size_t OFF_FB   = OFF_FA  + (size_t)S_*T_*4;           // S*T f32
static constexpr size_t OFF_BP   = OFF_FB  + (size_t)S_*T_*4;           // S*S u32       (16.8 MB)
static constexpr size_t OFF_BEST = OFF_BP  + (size_t)S_*S_*4;           // S int

// ---------------- K0: embedding gather + bf16 convert ----------------
__global__ void k_gather(const int* __restrict__ sent, const float* __restrict__ emb,
                         unsigned short* __restrict__ xs) {
  const int s = blockIdx.x, e = threadIdx.x;
  const int tok = sent[s];
  xs[(size_t)s * E_ + e] = f2bf(emb[(size_t)tok * E_ + e]);
}

// ---------------- K1: Wih weights -> bf16 (both directions) ----------------
__global__ void k_wconv(const float* __restrict__ wf, const float* __restrict__ wb,
                        unsigned short* __restrict__ dst) {
  const int i = blockIdx.x * blockDim.x + threadIdx.x;   // 0 .. 2*G*E-1
  const int n = G_ * E_;
  dst[i] = f2bf(i < n ? wf[i] : wb[i - n]);
}

// ---------------- K2: Gx[dir] = xs @ Wih[dir]^T + b[dir]  (WMMA bf16) ----------------
// A = xs (S x E bf16, row-major), B = Wih (G x E bf16, row-major == B^T, K contiguous).
// Each wave computes a 16x64 strip (1 M-tile x 4 N-tiles): the A fragment is loaded
// once per K-step and reused across 4 WMMAs (fetch:math ratio 2.5 loads/wmma).
// A/B lane layout (ISA 7.12.2, 16-bit 16x32): lane<16 -> row m=lane, K in {k0..k0+7, k0+16..k0+23}
//                                             lane>=16 -> row m=lane-16, K in {k0+8..15, k0+24..31}
__global__ void __launch_bounds__(256) k_gemm(const unsigned short* __restrict__ xs_u,
                                              const unsigned short* __restrict__ w_u,
                                              const float* __restrict__ bias_f,
                                              const float* __restrict__ bias_b,
                                              float* __restrict__ gx) {
  const int dir = blockIdx.y;
  const __bf16* xs = (const __bf16*)xs_u;
  const __bf16* w  = (const __bf16*)w_u + (size_t)dir * G_ * E_;
  const float* bias = dir ? bias_b : bias_f;
  float* out = gx + (size_t)dir * S_ * G_;

  const int wave = threadIdx.x >> 5;
  const int lane = threadIdx.x & 31;
  const int job  = blockIdx.x * 8 + wave;       // 0 .. 2047 (128 M-tiles x 16 N-strips)
  const int m0 = (job >> 4) << 4;               // M tile base
  const int n0 = (job & 15) << 6;               // 64-wide N strip base

  const int half = lane >> 4;                   // 0 | 1
  const int l15  = lane & 15;
  const int koff = half << 3;                   // 0 | 8

  const __bf16* arow = xs + (size_t)(m0 + l15) * E_;
  const __bf16* brow[4];
#pragma unroll
  for (int i = 0; i < 4; ++i) brow[i] = w + (size_t)(n0 + 16 * i + l15) * E_;

  v8f acc[4] = {{}, {}, {}, {}};
#pragma unroll
  for (int k0 = 0; k0 < E_; k0 += 32) {
    v8bf a0 = *(const v8bf*)(arow + k0 + koff);
    v8bf a1 = *(const v8bf*)(arow + k0 + koff + 16);
    v16bf a = __builtin_shufflevector(a0, a1, 0,1,2,3,4,5,6,7,8,9,10,11,12,13,14,15);
#pragma unroll
    for (int i = 0; i < 4; ++i) {
      v8bf b0 = *(const v8bf*)(brow[i] + k0 + koff);
      v8bf b1 = *(const v8bf*)(brow[i] + k0 + koff + 16);
      v16bf b = __builtin_shufflevector(b0, b1, 0,1,2,3,4,5,6,7,8,9,10,11,12,13,14,15);
      acc[i] = __builtin_amdgcn_wmma_f32_16x16x32_bf16(false, a, false, b, (short)0,
                                                       acc[i], false, false);
    }
  }

  // C/D layout: VGPR r, lanes 0-15 -> M=r, lanes 16-31 -> M=8+r; N = lane&15
  const int mbase = m0 + (half << 3);
#pragma unroll
  for (int i = 0; i < 4; ++i) {
    const int n = n0 + 16 * i + l15;
    const float bn = bias[n];
#pragma unroll
    for (int r = 0; r < 8; ++r)
      out[(size_t)(mbase + r) * G_ + n] = acc[i][r] + bn;
  }
}

// ---------------- K3: LSTM recurrence (one 1024-thread block per direction) ----------------
__global__ void __launch_bounds__(1024) k_lstm(const float* __restrict__ gx,
                                               const float* __restrict__ Whh_f,
                                               const float* __restrict__ Whh_b,
                                               const float* __restrict__ h0,
                                               const float* __restrict__ c0,
                                               float* __restrict__ h_all) {
  const int dir = blockIdx.x;
  const float* gxd  = gx + (size_t)dir * S_ * G_;
  const float* Whh  = dir ? Whh_b : Whh_f;
  float* hout = h_all + (size_t)dir * S_ * Hh_;

  __shared__ float h_s[Hh_];
  __shared__ float g_s[G_];
  const int tid = threadIdx.x;                // == gate row j

  float c_reg = 0.0f;
  if (tid < Hh_) { h_s[tid] = h0[dir * Hh_ + tid]; c_reg = c0[dir * Hh_ + tid]; }
  __syncthreads();

  const float4* w4 = (const float4*)(Whh + (size_t)tid * Hh_);
  for (int step = 0; step < S_; ++step) {
    const int t = dir ? (S_ - 1 - step) : step;
    // hide next step's only global dependency (gfx1250 global_prefetch_b8)
    if (step + 1 < S_) {
      const int tn = dir ? (S_ - 2 - step) : (step + 1);
      __builtin_prefetch(gxd + (size_t)tn * G_ + tid, 0, 1);
    }
    float acc = gxd[(size_t)t * G_ + tid];
    const float4* h4 = (const float4*)h_s;    // same-address LDS reads broadcast
#pragma unroll 8
    for (int k = 0; k < Hh_ / 4; ++k) {
      const float4 wv = w4[k];
      const float4 hv = h4[k];
      acc += wv.x * hv.x + wv.y * hv.y + wv.z * hv.z + wv.w * hv.w;
    }
    g_s[tid] = acc;
    __syncthreads();
    if (tid < Hh_) {
      const float ig = sigmf(g_s[tid]);
      const float fg = sigmf(g_s[Hh_ + tid]);
      const float gg = tanhf(g_s[2 * Hh_ + tid]);
      const float og = sigmf(g_s[3 * Hh_ + tid]);
      c_reg = fg * c_reg + ig * gg;
      const float h = og * tanhf(c_reg);
      h_s[tid] = h;
      hout[(size_t)t * Hh_ + tid] = h;
    }
    __syncthreads();
  }
}

// ---------------- K4: f_a = lstm_out@W1^T + b_tag, f_b = lstm_out@W2^T ----------------
__global__ void __launch_bounds__(32) k_tag(const float* __restrict__ h_all,
                                            const float* __restrict__ W_tag,
                                            const float* __restrict__ b_tag,
                                            float* __restrict__ fa, float* __restrict__ fb) {
  const int s = blockIdx.x;
  const int lane = threadIdx.x;               // 32 lanes
  const float* hf = h_all + (size_t)s * Hh_;
  const float* hb = h_all + (size_t)S_ * Hh_ + (size_t)s * Hh_;
  float hfv[8], hbv[8];
#pragma unroll
  for (int r = 0; r < 8; ++r) { hfv[r] = hf[lane + 32 * r]; hbv[r] = hb[lane + 32 * r]; }
#pragma unroll
  for (int tag = 0; tag < T_; ++tag) {
    const float* wt = W_tag + (size_t)tag * 1024;
    float aA = 0.0f, aB = 0.0f;
#pragma unroll
    for (int r = 0; r < 8; ++r) {
      const int k = lane + 32 * r;
      aA += hfv[r] * wt[k]       + hbv[r] * wt[256 + k];
      aB += hfv[r] * wt[512 + k] + hbv[r] * wt[768 + k];
    }
#pragma unroll
    for (int m = 16; m >= 1; m >>= 1) { aA += __shfl_xor(aA, m, 32); aB += __shfl_xor(aB, m, 32); }
    if (lane == 0) { fa[s * T_ + tag] = aA + b_tag[tag]; fb[s * T_ + tag] = aB; }
  }
}

// ---------------- K5: Viterbi forward (one lattice per lane) ----------------
__global__ void __launch_bounds__(256) k_vfwd(const float* __restrict__ fa,
                                              const float* __restrict__ fb,
                                              const float* __restrict__ trans,
                                              unsigned* __restrict__ bp,
                                              float* __restrict__ score_out,
                                              int* __restrict__ best_out) {
  const int s = blockIdx.x * blockDim.x + threadIdx.x;
  float tr[T_ * T_];
#pragma unroll
  for (int i = 0; i < T_ * T_; ++i) tr[i] = trans[i];
  float fbl[T_];
#pragma unroll
  for (int j = 0; j < T_; ++j) fbl[j] = fb[(size_t)s * T_ + j];
  float fv[T_];
#pragma unroll
  for (int j = 0; j < T_; ++j) fv[j] = (j == START_) ? 0.0f : NEG_;

  for (int t = 0; t < S_; ++t) {
    float fal[T_];
#pragma unroll
    for (int j = 0; j < T_; ++j) fal[j] = fa[(size_t)t * T_ + j];
    float nv[T_];
    unsigned w = 0;
#pragma unroll
    for (int j = 0; j < T_; ++j) {
      float best = fv[0] + tr[j * T_];
      int bi = 0;
#pragma unroll
      for (int k = 1; k < T_; ++k) {
        const float v = fv[k] + tr[j * T_ + k];
        if (v > best) { best = v; bi = k; }   // strict '>' keeps first max (jnp.argmax)
      }
      nv[j] = best + fal[j] + fbl[j];
      w |= (unsigned)bi << (3 * j);
    }
#pragma unroll
    for (int j = 0; j < T_; ++j) fv[j] = nv[j];
    bp[(size_t)t * S_ + s] = w;               // coalesced across lanes
  }
  float best = fv[0] + tr[STOP_ * T_];
  int bi = 0;
#pragma unroll
  for (int j = 1; j < T_; ++j) {
    const float v = fv[j] + tr[STOP_ * T_ + j];
    if (v > best) { best = v; bi = j; }
  }
  score_out[s] = best;                        // path_score
  best_out[s] = bi;
}

// ---------------- K6: Viterbi backtrack -> path[t][s] (coalesced) ----------------
__global__ void __launch_bounds__(256) k_vback(const unsigned* __restrict__ bp,
                                               const int* __restrict__ best,
                                               float* __restrict__ path_ts) {
  const int s = blockIdx.x * blockDim.x + threadIdx.x;
  int b = best[s];
  path_ts[(size_t)(S_ - 1) * S_ + s] = (float)b;
  for (int t = S_ - 1; t >= 1; --t) {
    const unsigned w = bp[(size_t)t * S_ + s];
    b = (int)((w >> (3 * b)) & 7u);
    path_ts[(size_t)(t - 1) * S_ + s] = (float)b;
  }
}

// ---------------- K7: transpose path[t][s] -> tag_seq[s][t] into d_out ----------------
__global__ void k_transpose(const float* __restrict__ path_ts, float* __restrict__ out_tags) {
  __shared__ float tile[32][33];
  const int x  = blockIdx.x * 32 + threadIdx.x;     // s (read)
  const int y0 = blockIdx.y * 32;                   // t
#pragma unroll
  for (int i = 0; i < 32; i += 8)
    tile[threadIdx.y + i][threadIdx.x] = path_ts[(size_t)(y0 + threadIdx.y + i) * S_ + x];
  __syncthreads();
  const int xo = blockIdx.y * 32 + threadIdx.x;     // t (write)
  const int yo = blockIdx.x * 32;                   // s
#pragma unroll
  for (int i = 0; i < 32; i += 8)
    out_tags[(size_t)(yo + threadIdx.y + i) * S_ + xo] = tile[threadIdx.x][threadIdx.y + i];
}

// ---------------- host entry ----------------
extern "C" void kernel_launch(void* const* d_in, const int* in_sizes, int n_in,
                              void* d_out, int out_size, void* d_ws, size_t ws_size,
                              hipStream_t stream) {
  (void)in_sizes; (void)n_in; (void)out_size; (void)ws_size;
  const int*   sentence = (const int*)  d_in[0];
  const float* emb      = (const float*)d_in[1];
  const float* Wih_f    = (const float*)d_in[2];
  const float* Whh_f    = (const float*)d_in[3];
  const float* b_f      = (const float*)d_in[4];
  const float* Wih_b    = (const float*)d_in[5];
  const float* Whh_b    = (const float*)d_in[6];
  const float* b_b      = (const float*)d_in[7];
  const float* W_tag    = (const float*)d_in[8];
  const float* b_tag    = (const float*)d_in[9];
  const float* trans    = (const float*)d_in[10];
  const float* h0       = (const float*)d_in[11];
  const float* c0       = (const float*)d_in[12];

  char* ws = (char*)d_ws;
  unsigned short* xs_bf   = (unsigned short*)(ws + OFF_XS);
  unsigned short* w_bf    = (unsigned short*)(ws + OFF_W);
  float*          gx      = (float*)(ws + OFF_GX);
  float*          h_all   = (float*)(ws + OFF_H);
  float*          fa      = (float*)(ws + OFF_FA);
  float*          fb      = (float*)(ws + OFF_FB);
  unsigned*       bp      = (unsigned*)(ws + OFF_BP);
  int*            best    = (int*)(ws + OFF_BEST);
  float*          path_ts = (float*)(ws + OFF_GX);   // Gx region is dead by then (16.8 MB)

  float* score_out = (float*)d_out;                  // [0, S)
  float* tags_out  = (float*)d_out + S_;             // [S, S + S*S)

  k_gather<<<S_, E_, 0, stream>>>(sentence, emb, xs_bf);
  k_wconv<<<(2 * G_ * E_) / 256, 256, 0, stream>>>(Wih_f, Wih_b, w_bf);
  k_gemm<<<dim3(256, 2), 256, 0, stream>>>(xs_bf, w_bf, b_f, b_b, gx);
  k_lstm<<<2, 1024, 0, stream>>>(gx, Whh_f, Whh_b, h0, c0, h_all);
  k_tag<<<S_, 32, 0, stream>>>(h_all, W_tag, b_tag, fa, fb);
  k_vfwd<<<S_ / 256, 256, 0, stream>>>(fa, fb, trans, bp, score_out, best);
  k_vback<<<S_ / 256, 256, 0, stream>>>(bp, best, path_ts);
  k_transpose<<<dim3(S_ / 32, S_ / 32), dim3(32, 8), 0, stream>>>(path_ts, tags_out);
}
